// RowClassifierSequenceModel_6691559047565
// MI455X (gfx1250) — matrix operand order repaired and checked by reading, not compiled, not run on hardware
//
#include <hip/hip_runtime.h>
#include <hip/hip_bf16.h>
#include <math.h>

typedef __attribute__((ext_vector_type(16))) _Float16 v16h;
typedef __attribute__((ext_vector_type(8)))  float    v8f;

#define BB 16
#define SS 512
#define TT 50
#define FF 15
#define EE 128
#define HH 256
#define CC 10
#define DD 143
#define KPAD 160
#define G4H 1024
#define MROWS (BB*SS)   /* 8192 */

__device__ __forceinline__ float sigf(float x) { return 1.0f / (1.0f + __expf(-x)); }

// 16-bit A-matrix 16x32 fragment: lane holds row M=lane&15; VGPR v (j=2v+h):
//   v<4 : K = 2v+h (+8 for lanes 16-31) ;  v>=4 : K = 16+2(v-4)+h (+8 for hi lanes)
__device__ __forceinline__ int kA(int j, int lane) {
  int v = j >> 1, h = j & 1;
  int base = (v < 4) ? (2 * v + h) : (16 + 2 * (v - 4) + h);
  return base + ((lane >= 16) ? 8 : 0);
}
// 16-bit B-matrix 32x16 fragment: lane holds col N=lane&15; same K order as A
__device__ __forceinline__ int kB(int j, int lane) {
  int v = j >> 1, h = j & 1;
  int base = (v < 4) ? (2 * v + h) : (16 + 2 * (v - 4) + h);
  return base + ((lane >= 16) ? 8 : 0);
}

// ---------------- weight fp32 -> fp16 with K padding ----------------
__global__ void k_f32_to_h_pad(const float* __restrict__ src, _Float16* __restrict__ dst,
                               int rows, int cols, int colsPad) {
  int idx = blockIdx.x * blockDim.x + threadIdx.x;
  if (idx >= rows * colsPad) return;
  int r = idx / colsPad, c = idx % colsPad;
  float v = (c < cols) ? src[r * cols + c] : 0.0f;
  dst[idx] = (_Float16)v;
}

// ---------------- embedding mean-pool + LayerNorm -> merged half (8192 x 160) ----------------
__global__ void k_embed_ln(const int* __restrict__ texts, const float* __restrict__ numerics,
                           const float* __restrict__ emb, const float* __restrict__ ln_g,
                           const float* __restrict__ ln_b, _Float16* __restrict__ mergedH) {
  __shared__ int   toks[TT];
  __shared__ float nums[FF];
  int row = blockIdx.x;                // b*S + s
  int tid = threadIdx.x;               // 0..159
  if (tid < TT) toks[tid] = texts[row * TT + tid];
  if (tid < FF) nums[tid] = numerics[row * FF + tid];
  __syncthreads();
  float val = 0.0f;
  if (tid < EE) {
    float sum = 0.0f, cnt = 0.0f;
    for (int t = 0; t < TT; ++t) {
      int tok = toks[t];
      if (tok != 0) { sum += emb[tok * EE + tid]; cnt += 1.0f; }
    }
    val = sum / fmaxf(cnt, 1e-9f);
  } else if (tid < EE + FF) {
    int f = tid - EE;
    float mu = 0.0f;
    for (int i = 0; i < FF; ++i) mu += nums[i];
    mu *= (1.0f / FF);
    float var = 0.0f;
    for (int i = 0; i < FF; ++i) { float d = nums[i] - mu; var += d * d; }
    var *= (1.0f / FF);
    val = (nums[f] - mu) * rsqrtf(var + 1e-5f) * ln_g[f] + ln_b[f];
  }
  mergedH[row * KPAD + tid] = (_Float16)val;
}

// ---------------- xg = merged @ W_ih^T + b   (M=8192, K=160, N=1024), WMMA f16 ----------------
__global__ __launch_bounds__(256) void k_xg_gemm(const _Float16* __restrict__ A,   // (8192,160)
                                                 const _Float16* __restrict__ Wh,  // (1024,160)
                                                 const float* __restrict__ bias,   // (1024)
                                                 float* __restrict__ out) {        // (8192,1024)
  int wave = blockIdx.x * (blockDim.x >> 5) + (threadIdx.x >> 5);
  int lane = threadIdx.x & 31;
  int mTile = wave >> 6;               // 512 M-tiles
  int nTile = wave & 63;               // 64 N-tiles
  int mBase = mTile * 16, nBase = nTile * 16;
  int laneLo = lane & 15;
  v8f acc = {};
#pragma unroll
  for (int kt = 0; kt < KPAD / 32; ++kt) {
    v16h a, b;
    int kb = kt * 32;
#pragma unroll
    for (int j = 0; j < 16; ++j) {
      a[j] = A[(mBase + laneLo) * KPAD + kb + kA(j, lane)];
      b[j] = Wh[(nBase + laneLo) * KPAD + kb + kB(j, lane)];
    }
    acc = __builtin_amdgcn_wmma_f32_16x16x32_f16(false, a, false, b, (short)0, acc, false, false);
  }
  int n = nBase + laneLo;
  float bv = bias[n];
#pragma unroll
  for (int i = 0; i < 8; ++i) {
    int m = mBase + i + ((lane >= 16) ? 8 : 0);
    out[m * G4H + n] = acc[i] + bv;
  }
}

// ---------------- sequential BiLSTM recurrence: 1 workgroup (32 waves) per direction ----------------
__global__ __launch_bounds__(1024) void k_lstm(const float* __restrict__ xgF, const float* __restrict__ xgB,
                                               const _Float16* __restrict__ WhhF, const _Float16* __restrict__ WhhB,
                                               const int* __restrict__ lengths, float* __restrict__ bilstm) {
  extern __shared__ char smem[];
  _Float16* hS = (_Float16*)smem;                          // 16 x 256 half = 8 KB
  float*    gS = (float*)(smem + BB * HH * sizeof(_Float16)); // 16 x 1024 f32 = 64 KB
  int dir = blockIdx.x;
  const float*    xg  = dir ? xgB  : xgF;
  const _Float16* Whh = dir ? WhhB : WhhF;
  int tid = threadIdx.x;
  int lane = tid & 31, wave = tid >> 5;
  int laneLo = lane & 15;
  for (int i = tid; i < BB * HH; i += 1024) hS[i] = (_Float16)0.0f;
  float cReg[4] = {0.f, 0.f, 0.f, 0.f};
  int   len[4];
#pragma unroll
  for (int i = 0; i < 4; ++i) { int e = i * 1024 + tid; len[i] = lengths[e >> 8]; }
  __syncthreads();

  for (int step = 0; step < SS; ++step) {
    int t = dir ? (SS - 1 - step) : step;
    // ---- g = xg[:, t, :] + h @ Whh^T  via WMMA (M=16 batch, K=256, N=1024) ----
#pragma unroll
    for (int w = 0; w < 2; ++w) {
      int nTile = wave * 2 + w;
      int n = nTile * 16 + laneLo;
      v8f acc;
#pragma unroll
      for (int i = 0; i < 8; ++i) {
        int brow = i + ((lane >= 16) ? 8 : 0);
        acc[i] = xg[(brow * SS + t) * G4H + n];
      }
#pragma unroll
      for (int kt = 0; kt < HH / 32; ++kt) {
        v16h a, b;
        int kb = kt * 32;
#pragma unroll
        for (int j = 0; j < 16; ++j) {
          a[j] = hS[laneLo * HH + kb + kA(j, lane)];
          b[j] = Whh[n * HH + kb + kB(j, lane)];
        }
        acc = __builtin_amdgcn_wmma_f32_16x16x32_f16(false, a, false, b, (short)0, acc, false, false);
      }
#pragma unroll
      for (int i = 0; i < 8; ++i) {
        int brow = i + ((lane >= 16) ? 8 : 0);
        gS[brow * G4H + n] = acc[i];
      }
    }
    __syncthreads();
    // ---- gate nonlinearity + state update (4096 cells / 1024 threads) ----
#pragma unroll
    for (int i = 0; i < 4; ++i) {
      int e = i * 1024 + tid;
      int brow = e >> 8, hh = e & 255;
      float gi = gS[brow * G4H + hh];
      float gf = gS[brow * G4H + 256 + hh];
      float gg = gS[brow * G4H + 512 + hh];
      float go = gS[brow * G4H + 768 + hh];
      bool  msk = (t < len[i]);
      float c  = cReg[i];
      float cn = sigf(gf) * c + sigf(gi) * tanhf(gg);
      float hn = sigf(go) * tanhf(cn);
      float hprev = (float)hS[brow * HH + hh];
      float hcur  = msk ? hn : hprev;
      cReg[i] = msk ? cn : c;
      bilstm[(brow * SS + t) * (2 * HH) + dir * HH + hh] = msk ? hcur : 0.0f;
      hS[brow * HH + hh] = (_Float16)hcur;
    }
    __syncthreads();
  }
}

// ---------------- emissions = bilstm @ fc_W^T + fc_b  (tiny) ----------------
__global__ void k_emis(const float* __restrict__ bilstm, const float* __restrict__ fcW,
                       const float* __restrict__ fcb, float* __restrict__ emis) {
  int row = blockIdx.x;
  int c = threadIdx.x;
  if (c >= CC) return;
  float acc = fcb[c];
  const float* hrow = bilstm + (size_t)row * (2 * HH);
  const float* wrow = fcW + c * (2 * HH);
  for (int k = 0; k < 2 * HH; ++k) acc += hrow[k] * wrow[k];
  emis[row * CC + c] = acc;
}

// ---------------- CRF: gold score + forward logZ -> -mean(llh) ----------------
__global__ __launch_bounds__(256) void k_crf(const float* __restrict__ emis, const int* __restrict__ targets,
                                             const int* __restrict__ lengths, const float* __restrict__ start,
                                             const float* __restrict__ endv, const float* __restrict__ trans,
                                             float* __restrict__ out) {
  __shared__ float alpha[BB][CC];
  __shared__ float trS[CC * CC];
  __shared__ float scoreS[BB];
  __shared__ float res[BB];
  int tid = threadIdx.x;
  int b = tid >> 4, j = tid & 15;
  if (tid < CC * CC) trS[tid] = trans[tid];
  if (j < CC) alpha[b][j] = start[j] + emis[(b * SS) * CC + j];
  __syncthreads();
  if (j == 0) {
    int L = lengths[b];
    int y0 = targets[b * SS];
    float sc = start[y0] + emis[(b * SS) * CC + y0];
    for (int t = 1; t < SS; ++t) {
      if (t < L) {
        int yp = targets[b * SS + t - 1];
        int yt = targets[b * SS + t];
        sc += trS[yp * CC + yt] + emis[(b * SS + t) * CC + yt];
      }
    }
    sc += endv[targets[b * SS + L - 1]];
    scoreS[b] = sc;
  }
  for (int t = 1; t < SS; ++t) {
    float nxt = 0.0f;
    bool act = (j < CC);
    if (act) {
      float mx = -1e30f;
#pragma unroll
      for (int i = 0; i < CC; ++i) mx = fmaxf(mx, alpha[b][i] + trS[i * CC + j]);
      float s = 0.0f;
#pragma unroll
      for (int i = 0; i < CC; ++i) s += __expf(alpha[b][i] + trS[i * CC + j] - mx);
      nxt = mx + __logf(s) + emis[(b * SS + t) * CC + j];
    }
    __syncthreads();
    if (act && t < lengths[b]) alpha[b][j] = nxt;
    __syncthreads();
  }
  if (j == 0) {
    float mx = -1e30f;
    for (int i = 0; i < CC; ++i) mx = fmaxf(mx, alpha[b][i] + endv[i]);
    float s = 0.0f;
    for (int i = 0; i < CC; ++i) s += __expf(alpha[b][i] + endv[i] - mx);
    res[b] = scoreS[b] - (mx + __logf(s));
  }
  __syncthreads();
  if (tid == 0) {
    float acc = 0.0f;
    for (int i = 0; i < BB; ++i) acc += res[i];
    out[0] = -acc * (1.0f / BB);
  }
}

extern "C" void kernel_launch(void* const* d_in, const int* in_sizes, int n_in,
                              void* d_out, int out_size, void* d_ws, size_t ws_size,
                              hipStream_t stream) {
  const int*   texts     = (const int*)  d_in[0];
  const float* numerics  = (const float*)d_in[1];
  const int*   lengths   = (const int*)  d_in[2];
  const int*   targets   = (const int*)  d_in[3];
  const float* emb       = (const float*)d_in[4];
  const float* ln_g      = (const float*)d_in[5];
  const float* ln_b      = (const float*)d_in[6];
  const float* W_ih_f    = (const float*)d_in[7];
  const float* W_hh_f    = (const float*)d_in[8];
  const float* b_f       = (const float*)d_in[9];
  const float* W_ih_b    = (const float*)d_in[10];
  const float* W_hh_b    = (const float*)d_in[11];
  const float* b_b       = (const float*)d_in[12];
  const float* fc_W      = (const float*)d_in[13];
  const float* fc_b      = (const float*)d_in[14];
  const float* crf_start = (const float*)d_in[15];
  const float* crf_end   = (const float*)d_in[16];
  const float* crf_trans = (const float*)d_in[17];
  float* out = (float*)d_out;

  size_t off = 0;
  auto carve = [&](size_t bytes) -> void* {
    off = (off + 255) & ~(size_t)255;
    void* p = (char*)d_ws + off;
    off += bytes;
    return p;
  };
  _Float16* mergedH = (_Float16*)carve((size_t)MROWS * KPAD * 2);
  _Float16* WihFh   = (_Float16*)carve((size_t)G4H * KPAD * 2);
  _Float16* WihBh   = (_Float16*)carve((size_t)G4H * KPAD * 2);
  _Float16* WhhFh   = (_Float16*)carve((size_t)G4H * HH * 2);
  _Float16* WhhBh   = (_Float16*)carve((size_t)G4H * HH * 2);
  float*    xgF     = (float*)carve((size_t)MROWS * G4H * 4);
  float*    xgB     = (float*)carve((size_t)MROWS * G4H * 4);
  float*    bilstm  = (float*)carve((size_t)MROWS * 2 * HH * 4);
  float*    emis    = (float*)carve((size_t)MROWS * CC * 4);

  int tot = G4H * KPAD;
  k_f32_to_h_pad<<<(tot + 255) / 256, 256, 0, stream>>>(W_ih_f, WihFh, G4H, DD, KPAD);
  k_f32_to_h_pad<<<(tot + 255) / 256, 256, 0, stream>>>(W_ih_b, WihBh, G4H, DD, KPAD);
  tot = G4H * HH;
  k_f32_to_h_pad<<<(tot + 255) / 256, 256, 0, stream>>>(W_hh_f, WhhFh, G4H, HH, HH);
  k_f32_to_h_pad<<<(tot + 255) / 256, 256, 0, stream>>>(W_hh_b, WhhBh, G4H, HH, HH);

  k_embed_ln<<<MROWS, KPAD, 0, stream>>>(texts, numerics, emb, ln_g, ln_b, mergedH);

  // 512 M-tiles x 64 N-tiles = 32768 waves, 8 waves / block
  k_xg_gemm<<<32768 / 8, 256, 0, stream>>>(mergedH, WihFh, b_f, xgF);
  k_xg_gemm<<<32768 / 8, 256, 0, stream>>>(mergedH, WihBh, b_b, xgB);

  k_lstm<<<2, 1024, BB * HH * sizeof(_Float16) + BB * G4H * sizeof(float), stream>>>(
      xgF, xgB, WhhFh, WhhBh, lengths, bilstm);

  k_emis<<<MROWS, 32, 0, stream>>>(bilstm, fc_W, fc_b, emis);
  k_crf<<<1, 256, 0, stream>>>(emis, targets, lengths, crf_start, crf_end, crf_trans, out);
}
